// GAEModel_79834852098581
// MI455X (gfx1250) — matrix operand order repaired and checked by reading, not compiled, not run on hardware
//
#include <hip/hip_runtime.h>
#include <hip/hip_bf16.h>

#define N_NODES 4096
#define N_EDGES 32768
#define NODE_DIM 32
#define HIDDEN 64
#define EDGE_DIM 32
#define NEG_SLOPE 0.2f

typedef __attribute__((ext_vector_type(16))) __bf16        v16bf;
typedef __attribute__((ext_vector_type(8)))  float         v8f;
typedef __attribute__((ext_vector_type(8)))  unsigned short ush8;

union ABop { v16bf v; ush8 u[2]; };

// ---------------------------------------------------------------- utilities

__global__ __launch_bounds__(256) void zero_f32(float* __restrict__ p, int n) {
    int i = blockIdx.x * 256 + threadIdx.x;
    if (i < n) p[i] = 0.0f;
}

// round-to-nearest-even f32 -> bf16
__global__ __launch_bounds__(256) void f32_to_bf16(const float* __restrict__ s,
                                                   unsigned short* __restrict__ d, int n) {
    int i = blockIdx.x * 256 + threadIdx.x;
    if (i < n) {
        unsigned u = __float_as_uint(s[i]);
        unsigned r = u + 0x7FFFu + ((u >> 16) & 1u);
        d[i] = (unsigned short)(r >> 16);
    }
}

// ------------------------------------------------- fused NNConv edge kernel
//
// Per block: 64 edges (4 tiles of 16), 256 threads = 8 waves.
// Wave w: obase = (w&3)*16 output columns, iphase = w>>2 covers half the i's.
// For each i: 4 independent bf16 WMMAs (one per edge-tile) are issued
// back-to-back so WMMA->VALU hazard slots are filled by co-executing WMMAs:
//   D(16x16) = EA_tile(16x32) @ nnW.T_cols(32x16) + bias   (bias in C operand)
// then leaky_relu in registers and contract against h[src[e]][i] into macc.
// The (E x IN_DIM*64) intermediate never leaves VGPRs.
template <int IN_DIM>
__global__ __launch_bounds__(256)
void edge_conv(const unsigned short* __restrict__ eaBf,   // E x 32 bf16
               const unsigned short* __restrict__ wBf,    // (IN_DIM*64) x 32 bf16
               const float* __restrict__ nnb,             // IN_DIM*64
               const float* __restrict__ hsrc,            // N x IN_DIM f32
               const int*   __restrict__ srcIdx,
               const int*   __restrict__ dstIdx,
               float* __restrict__ sOut,                  // N x 64 (atomic accum)
               float* __restrict__ cntOut)                // N (atomic accum) or null
{
    __shared__ int   sL[64];
    __shared__ int   dL[64];
    __shared__ float hsS[64 * IN_DIM];
    __shared__ float msg[64 * 65];     // padded to kill bank conflicts

    const int t  = threadIdx.x;
    const int eb = blockIdx.x * 64;

    if (t < 64) { sL[t] = srcIdx[eb + t]; dL[t] = dstIdx[eb + t]; }
    __syncthreads();

    // stage h[src[e]] rows
    for (int idx = t; idx < 64 * IN_DIM; idx += 256) {
        int e = idx / IN_DIM, i = idx % IN_DIM;
        hsS[idx] = hsrc[(size_t)sL[e] * IN_DIM + i];
    }
    __syncthreads();

    const int wave   = t >> 5;
    const int lane   = t & 31;
    const int n      = lane & 15;
    const int hh     = lane >> 4;      // lane half
    const int obase  = (wave & 3) * 16;
    const int iphase = wave >> 2;

    // A operands: 16x32 bf16 edge_attr tiles (ISA 16-bit A layout)
    ABop a[4];
#pragma unroll
    for (int et = 0; et < 4; ++et) {
        const unsigned short* Ab = eaBf + (size_t)(eb + et * 16 + n) * 32;
        a[et].u[0] = *(const ush8*)(Ab + 8 * hh);
        a[et].u[1] = *(const ush8*)(Ab + 16 + 8 * hh);
    }

    v8f macc[4];
#pragma unroll
    for (int et = 0; et < 4; ++et)
#pragma unroll
        for (int r = 0; r < 8; ++r) macc[et][r] = 0.0f;

    constexpr int NI = IN_DIM / 2;     // i's per wave
    const int i0 = iphase * NI;

    // double-buffered B operand (32x16 bf16: lanes 0-15 K=0..7,16..23, 16-31 K=8..15,24..31)
    ABop  bcur;
    float biascur;
    {
        const int j = i0 * 64 + obase + n;
        const unsigned short* Bp = wBf + (size_t)j * 32 + 8 * hh;
        bcur.u[0] = *(const ush8*)(Bp);
        bcur.u[1] = *(const ush8*)(Bp + 16);
        biascur   = nnb[j];
    }

    for (int it = 0; it < NI; ++it) {
        const int i = i0 + it;

        // prefetch next B + bias while this iteration computes
        ABop  bnxt;
        float biasnxt = 0.0f;
        if (it + 1 < NI) {
            const int jn = (i + 1) * 64 + obase + n;
            const unsigned short* Bp = wBf + (size_t)jn * 32 + 8 * hh;
            bnxt.u[0] = *(const ush8*)(Bp);
            bnxt.u[1] = *(const ush8*)(Bp + 16);
            biasnxt   = nnb[jn];
        }

        v8f c;
#pragma unroll
        for (int r = 0; r < 8; ++r) c[r] = biascur;

        // issue all 4 independent WMMAs back-to-back (hazard slots co-execute)
        v8f d[4];
#pragma unroll
        for (int et = 0; et < 4; ++et)
            d[et] = __builtin_amdgcn_wmma_f32_16x16x32_bf16(
                false, a[et].v, false, bcur.v, (short)0, c, false, false);

        // epilogue: leaky_relu + contraction against h[src][i]
#pragma unroll
        for (int et = 0; et < 4; ++et) {
#pragma unroll
            for (int r = 0; r < 8; ++r) {
                float v  = d[et][r];
                v        = (v > 0.0f) ? v : NEG_SLOPE * v;
                float hv = hsS[(et * 16 + r + 8 * hh) * IN_DIM + i];
                macc[et][r] = fmaf(hv, v, macc[et][r]);
            }
        }

        bcur    = bnxt;
        biascur = biasnxt;
    }

    // combine the two i-phases per output range (write, then add)
    if (iphase == 0) {
#pragma unroll
        for (int et = 0; et < 4; ++et)
#pragma unroll
            for (int r = 0; r < 8; ++r)
                msg[(et * 16 + r + 8 * hh) * 65 + obase + n] = macc[et][r];
    }
    __syncthreads();
    if (iphase == 1) {
#pragma unroll
        for (int et = 0; et < 4; ++et)
#pragma unroll
            for (int r = 0; r < 8; ++r)
                msg[(et * 16 + r + 8 * hh) * 65 + obase + n] += macc[et][r];
    }
    __syncthreads();

    // scatter to destination nodes
    for (int idx = t; idx < 64 * 64; idx += 256) {
        int e = idx >> 6, o = idx & 63;
        atomicAdd(&sOut[(size_t)dL[e] * 64 + o], msg[e * 65 + o]);
    }
    if (cntOut != nullptr && t < 64) atomicAdd(&cntOut[dL[t]], 1.0f);
}

// ------------------------------------------- node finish: layer 1 (+ relu)
__global__ __launch_bounds__(256)
void node_finish1(const float* __restrict__ s1, const float* __restrict__ cnt,
                  const float* __restrict__ x,  const float* __restrict__ root1,
                  const float* __restrict__ bias1, float* __restrict__ h1) {
    int idx = blockIdx.x * 256 + threadIdx.x;          // N*64 threads
    if (idx >= N_NODES * HIDDEN) return;
    int nd = idx >> 6, o = idx & 63;
    float c = cnt[nd]; c = (c > 1.0f) ? c : 1.0f;
    float acc = bias1[o] + s1[idx] / c;
    const float* xr = x + (size_t)nd * NODE_DIM;
#pragma unroll
    for (int i = 0; i < NODE_DIM; ++i) acc = fmaf(xr[i], root1[i * 64 + o], acc);
    h1[idx] = (acc > 0.0f) ? acc : 0.0f;
}

// --------------- node finish layer 2 + decoder projections a[n], b[n]
__global__ __launch_bounds__(64)
void node_finish2_ab(const float* __restrict__ s2, const float* __restrict__ cnt,
                     const float* __restrict__ h1, const float* __restrict__ root2,
                     const float* __restrict__ bias2, const float* __restrict__ linW,
                     float* __restrict__ aOut, float* __restrict__ bOut) {
    int nd = blockIdx.x;
    int o  = threadIdx.x;                               // 64 threads
    float c = cnt[nd]; c = (c > 1.0f) ? c : 1.0f;
    float acc = bias2[o] + s2[(size_t)nd * 64 + o] / c;
    const float* hr = h1 + (size_t)nd * 64;
#pragma unroll
    for (int i = 0; i < HIDDEN; ++i) acc = fmaf(hr[i], root2[i * 64 + o], acc);
    float hv = (acc > 0.0f) ? acc : 0.0f;               // relu

    __shared__ float ra[64], rb[64];
    ra[o] = hv * linW[o];
    rb[o] = hv * linW[64 + o];
    __syncthreads();
    if (o == 0) {
        float sa = 0.0f, sb = 0.0f;
        for (int i = 0; i < 64; ++i) { sa += ra[i]; sb += rb[i]; }
        aOut[nd] = sa;
        bOut[nd] = sb;
    }
}

// ------------------------------------------------ decoder: 64 MB writeout
__global__ __launch_bounds__(256)
void decoder(const float* __restrict__ aArr, const float* __restrict__ bArr,
             const float* __restrict__ linb, float* __restrict__ out) {
    int i  = blockIdx.x;                                // row
    int j  = blockIdx.y * 1024 + threadIdx.x * 4;       // col (4 per thread)
    float base = aArr[i] + linb[0];
    float4 bv = *(const float4*)(bArr + j);
    float4 r;
    r.x = 1.0f / (1.0f + __expf(-(base + bv.x)));
    r.y = 1.0f / (1.0f + __expf(-(base + bv.y)));
    r.z = 1.0f / (1.0f + __expf(-(base + bv.z)));
    r.w = 1.0f / (1.0f + __expf(-(base + bv.w)));
    *(float4*)(out + (size_t)i * N_NODES + j) = r;
}

// ---------------------------------------------------------------- launcher
extern "C" void kernel_launch(void* const* d_in, const int* in_sizes, int n_in,
                              void* d_out, int out_size, void* d_ws, size_t ws_size,
                              hipStream_t stream) {
    (void)in_sizes; (void)n_in; (void)out_size; (void)ws_size;

    const float* x     = (const float*)d_in[0];
    const int*   ei    = (const int*)  d_in[1];
    const float* ea    = (const float*)d_in[2];
    const float* nn1W  = (const float*)d_in[3];
    const float* nn1b  = (const float*)d_in[4];
    const float* root1 = (const float*)d_in[5];
    const float* bias1 = (const float*)d_in[6];
    const float* nn2W  = (const float*)d_in[7];
    const float* nn2b  = (const float*)d_in[8];
    const float* root2 = (const float*)d_in[9];
    const float* bias2 = (const float*)d_in[10];
    const float* linW  = (const float*)d_in[11];
    const float* linb  = (const float*)d_in[12];
    const int* src = ei;
    const int* dst = ei + N_EDGES;

    // workspace layout (all 16B-aligned)
    float* s1   = (float*)d_ws;                    // N*64
    float* s2   = s1  + N_NODES * 64;              // N*64
    float* cnt  = s2  + N_NODES * 64;              // N
    float* h1   = cnt + N_NODES;                   // N*64
    float* aArr = h1  + N_NODES * 64;              // N
    float* bArr = aArr + N_NODES;                  // N
    unsigned short* eaBf = (unsigned short*)(bArr + N_NODES);   // E*32
    unsigned short* w1Bf = eaBf + (size_t)N_EDGES * 32;         // 2048*32
    unsigned short* w2Bf = w1Bf + 2048 * 32;                    // 4096*32

    // 1) zero accumulators (s1, s2, cnt are contiguous)
    {
        int nz = N_NODES * 64 * 2 + N_NODES;
        zero_f32<<<(nz + 255) / 256, 256, 0, stream>>>(s1, nz);
    }
    // 2) bf16 operand conversion
    f32_to_bf16<<<(N_EDGES * 32 + 255) / 256, 256, 0, stream>>>(ea,   eaBf, N_EDGES * 32);
    f32_to_bf16<<<(2048 * 32 + 255) / 256,    256, 0, stream>>>(nn1W, w1Bf, 2048 * 32);
    f32_to_bf16<<<(4096 * 32 + 255) / 256,    256, 0, stream>>>(nn2W, w2Bf, 4096 * 32);

    // 3) layer 1: fused edge-NN GEMM (WMMA bf16) + message + scatter + degree
    edge_conv<NODE_DIM><<<N_EDGES / 64, 256, 0, stream>>>(
        eaBf, w1Bf, nn1b, x, src, dst, s1, cnt);

    // 4) layer 1 node finish (mean-agg + root + bias + relu)
    node_finish1<<<(N_NODES * 64) / 256, 256, 0, stream>>>(s1, cnt, x, root1, bias1, h1);

    // 5) layer 2: fused edge-NN GEMM (WMMA bf16) + message + scatter
    edge_conv<HIDDEN><<<N_EDGES / 64, 256, 0, stream>>>(
        eaBf, w2Bf, nn2b, h1, src, dst, s2, nullptr);

    // 6) layer 2 node finish + decoder projections a[n] = h·W1, b[n] = h·W2
    node_finish2_ab<<<N_NODES, 64, 0, stream>>>(s2, cnt, h1, root2, bias2, linW, aArr, bArr);

    // 7) N x N sigmoid writeout (bandwidth-bound: 64 MB)
    decoder<<<dim3(N_NODES, 4), 256, 0, stream>>>(aArr, bArr, linb, (float*)d_out);
}
